// AnchorModule_62388694941793
// MI455X (gfx1250) — matrix operand order repaired and checked by reading, not compiled, not run on hardware
//
#include <hip/hip_runtime.h>

// ---------------------------------------------------------------------------
// CDNA5 / gfx1250 implementation of the anchor-voxel-LSTM pipeline.
// wave32, WMMA f32_16x16x32_f16 for the PointNet MLP and the conv1 im2col
// GEMM, GLOBAL_LOAD_ASYNC_TO_LDS_B128 for LDS staging of pre-converted
// f16 weights.
// ---------------------------------------------------------------------------

typedef __attribute__((ext_vector_type(16))) _Float16 v16h;
typedef __attribute__((ext_vector_type(8)))  _Float16 v8h;
typedef __attribute__((ext_vector_type(8)))  float    v8f;

#define BTF     128          // batch*length
#define NANCH   81
#define NGRP    (BTF*NANCH)  // 10368
#define NSK     32           // neighbors per anchor
#define NPTS    8192
#define CIN_RAW 35
#define EPSBN   1e-5f

// --- gfx1250 async global->LDS copy (ASYNCcnt), CDNA5 ISA ---
__device__ __forceinline__ void async_b128(const void* gsrc, void* lds_dst)
{
  // LDS destination address comes from the vdst VGPR; generic shared ptr low 32 bits = LDS offset
  asm volatile("global_load_async_to_lds_b128 %0, %1, off"
               :
               : "v"((unsigned)(uintptr_t)lds_dst),
                 "v"((unsigned long long)(uintptr_t)gsrc)
               : "memory");
}
__device__ __forceinline__ void wait_async0()
{
  asm volatile("s_wait_asynccnt 0x0" ::: "memory");
}

// ---------------------------------------------------------------------------
// Kernel 0: one-shot f32 -> f16 weight conversion into workspace.
// w1h: 32x64 (K 38->64 zero-pad), w2h: 48x32, w3h: 64x64 (K 48->64 pad),
// vx1h: 96x1728 (conv1 im2col B panel).
// ---------------------------------------------------------------------------
__global__ void prep_kernel(const float* __restrict__ w1, const float* __restrict__ w2,
                            const float* __restrict__ w3, const float* __restrict__ vxw1,
                            _Float16* __restrict__ w1h, _Float16* __restrict__ w2h,
                            _Float16* __restrict__ w3h, _Float16* __restrict__ vx1h)
{
  const int t      = blockIdx.x * blockDim.x + threadIdx.x;
  const int stride = gridDim.x * blockDim.x;
  for (int e = t; e < 32 * 64; e += stride) {
    const int oc = e >> 6, k = e & 63;
    w1h[e] = (_Float16)((k < 38) ? w1[oc * 38 + k] : 0.f);
  }
  for (int e = t; e < 48 * 32; e += stride) w2h[e] = (_Float16)w2[e];
  for (int e = t; e < 64 * 64; e += stride) {
    const int oc = e >> 6, k = e & 63;
    w3h[e] = (_Float16)((k < 48) ? w3[oc * 48 + k] : 0.f);
  }
  for (int e = t; e < 96 * 1728; e += stride) vx1h[e] = (_Float16)vxw1[e];
}

// ---------------------------------------------------------------------------
// Kernel 1: squared-distance + top-32 selection per (frame, anchor).
// 256 threads; each thread caches 32 distances in VGPRs, 32 rounds of
// block-wide argmin (tie-break on lower point index, matching lax.top_k).
// ---------------------------------------------------------------------------
__global__ void topk_kernel(const float* __restrict__ x,
                            const float* __restrict__ g_loc,
                            int* __restrict__ idx_out)
{
  const int g   = blockIdx.x;
  const int bt  = g / NANCH;
  const int a   = g - bt * NANCH;
  const int tid = threadIdx.x;

  const int ix = a % 3, iy = (a / 3) % 3, iz = a / 9;
  const float ax = -2.0f + ix * 1.6f + g_loc[bt * 2 + 0];
  const float ay = -2.0f + iy * 3.0f + g_loc[bt * 2 + 1];
  const float az =  2.0f + iz * 0.3f;

  const float* xb = x + (size_t)bt * NPTS * CIN_RAW;

  float d[32];
  unsigned mask = 0u;
#pragma unroll
  for (int i = 0; i < 32; ++i) {
    const int p = tid + (i << 8);
    const float* pp = xb + (size_t)p * CIN_RAW;
    if (i + 1 < 32)  // stream next row through gfx1250 global_prefetch_b8
      __builtin_prefetch(xb + (size_t)(tid + ((i + 1) << 8)) * CIN_RAW, 0, 1);
    const float dx = pp[0] - ax, dy = pp[1] - ay, dz = pp[2] - az;
    d[i] = dx * dx + dy * dy + dz * dz;
  }

  __shared__ float sd[256];
  __shared__ int   si[256];

  for (int r = 0; r < NSK; ++r) {
    float best = 3.4e38f;
    int   bi   = 0x7fffffff;
#pragma unroll
    for (int i = 0; i < 32; ++i) {
      if (!((mask >> i) & 1u)) {
        const int p = tid + (i << 8);
        const float di = d[i];
        if (di < best || (di == best && p < bi)) { best = di; bi = p; }
      }
    }
    sd[tid] = best; si[tid] = bi;
    __syncthreads();
    for (int s = 128; s > 0; s >>= 1) {
      if (tid < s) {
        const float od = sd[tid + s];
        const int   oi = si[tid + s];
        if (od < sd[tid] || (od == sd[tid] && oi < si[tid])) { sd[tid] = od; si[tid] = oi; }
      }
      __syncthreads();
    }
    const int win = si[0];
    if (tid == 0) idx_out[(size_t)g * NSK + r] = win;
    if ((win & 255) == tid) mask |= 1u << (win >> 8);  // owner retires the winner
    __syncthreads();
  }
}

// ---------------------------------------------------------------------------
// WMMA helpers (A: 16x32 f16, k = 16*(lane/16) + s; B: 32x16, lane = out column).
// ---------------------------------------------------------------------------
__device__ __forceinline__ v16h load_afrag(const _Float16* __restrict__ row,
                                           int kc, int lane)
{
  const int ksub = (lane & 16) ? 16 : 0;
  return *(const v16h*)(row + kc + ksub);               // K = kc + {0..15}/{16..31}
}

// One PointNet layer: D(32 x Cout) = relu(alpha * (A(32 x K) @ W^T) + beta),
// activations stored f16 (stride 64) for the next layer. 4 waves share tiles.
__device__ __forceinline__ void mlp_layer(const _Float16* __restrict__ Ain,
                                          const _Float16* __restrict__ W,
                                          int Kpad, int Cout,
                                          const float* __restrict__ alpha,
                                          const float* __restrict__ beta,
                                          _Float16* __restrict__ Aout,
                                          int wave, int lane)
{
  const int tiles = (Cout >> 4) * 2;  // nMt = 2 (32 points)
  for (int t = wave; t < tiles; t += 4) {
    const int mt = t & 1, nt = t >> 1;
    v8f acc = {};
    const _Float16* arow = Ain + (mt * 16 + (lane & 15)) * 64;
    const _Float16* brow = W + (nt * 16 + (lane & 15)) * Kpad + ((lane >> 4) << 4);
    for (int kc = 0; kc < Kpad; kc += 32) {
      const v16h av = load_afrag(arow, kc, lane);
      const v16h bv = *(const v16h*)(brow + kc);
      acc = __builtin_amdgcn_wmma_f32_16x16x32_f16(false, av, false, bv,
                                                   (short)0, acc, false, false);
    }
    const int o = nt * 16 + (lane & 15);
    const float al = alpha[o], be = beta[o];
    const int mbase = mt * 16 + ((lane < 16) ? 0 : 8);
#pragma unroll
    for (int r = 0; r < 8; ++r) {
      const float y = acc[r] * al + be;
      Aout[(mbase + r) * 64 + o] = (_Float16)(y > 0.f ? y : 0.f);
    }
  }
}

struct PNP {
  const float *b1, *g1, *be1, *m1, *v1;
  const float *b2, *g2, *be2, *m2, *v2;
  const float *b3, *g3, *be3, *m3, *v3;
  const float *aw, *ab;
};

// ---------------------------------------------------------------------------
// Kernel 2: gather neighbors, 3-layer PointNet (WMMA), attention softmax,
// weighted pooling. Weights staged into LDS via async global->LDS b128.
// 128 threads = 4 waves per group.
// ---------------------------------------------------------------------------
__global__ void pointnet_kernel(const float* __restrict__ x,
                                const float* __restrict__ g_loc,
                                const int* __restrict__ idx,
                                const _Float16* __restrict__ w1h,
                                const _Float16* __restrict__ w2h,
                                const _Float16* __restrict__ w3h,
                                PNP p,
                                float* __restrict__ attn_out,
                                float* __restrict__ voxel)
{
  const int g    = blockIdx.x;
  const int bt   = g / NANCH;
  const int a    = g - bt * NANCH;
  const int tid  = threadIdx.x;
  const int wave = tid >> 5, lane = tid & 31;

  __shared__ __align__(32) _Float16 A0[32 * 64], A1[32 * 64], A2[32 * 64];
  __shared__ __align__(32) _Float16 W1s[32 * 64], W2s[48 * 32], W3s[64 * 64];
  __shared__ float al1[32], bt1[32], al2[48], bt2[48], al3[64], bt3[64];
  __shared__ float wat[64];
  __shared__ float lg[32], wsm[32];
  __shared__ int   pidx[32];

  const int ix = a % 3, iy = (a / 3) % 3, iz = a / 9;
  float anc[3];
  anc[0] = -2.0f + ix * 1.6f + g_loc[bt * 2 + 0];
  anc[1] = -2.0f + iy * 3.0f + g_loc[bt * 2 + 1];
  anc[2] =  2.0f + iz * 0.3f;

  if (tid < 32) pidx[tid] = idx[(size_t)g * NSK + tid];

  // pre-converted f16 weights -> LDS via ASYNCcnt-tracked DMA (b128 units)
  for (int u = tid; u < 256; u += 128) async_b128(w1h + u * 8, W1s + u * 8);
  for (int u = tid; u < 192; u += 128) async_b128(w2h + u * 8, W2s + u * 8);
  for (int u = tid; u < 512; u += 128) async_b128(w3h + u * 8, W3s + u * 8);

  // fused bias+BN: y = acc*alpha + beta
  if (tid < 32) { const float s = p.g1[tid] * rsqrtf(p.v1[tid] + EPSBN);
                  al1[tid] = s; bt1[tid] = (p.b1[tid] - p.m1[tid]) * s + p.be1[tid]; }
  if (tid < 48) { const float s = p.g2[tid] * rsqrtf(p.v2[tid] + EPSBN);
                  al2[tid] = s; bt2[tid] = (p.b2[tid] - p.m2[tid]) * s + p.be2[tid]; }
  if (tid < 64) { const float s = p.g3[tid] * rsqrtf(p.v3[tid] + EPSBN);
                  al3[tid] = s; bt3[tid] = (p.b3[tid] - p.m3[tid]) * s + p.be3[tid];
                  wat[tid] = p.aw[tid]; }
  for (int e = tid; e < 32 * 64; e += 128) A2[e] = (_Float16)0.f;  // pad cols 48..63
  wait_async0();
  __syncthreads();

  // gather: new_points = [anchor(3), xyz-anchor(3), pts(32)], pad K 38 -> 64
  for (int e = tid; e < 32 * 64; e += 128) {
    const int n = e >> 6, c = e & 63;
    float v = 0.f;
    if (c < 38) {
      const float* pr = x + ((size_t)bt * NPTS + pidx[n]) * CIN_RAW;
      if (c < 3)      v = anc[c];
      else if (c < 6) v = pr[c - 3] - anc[c - 3];
      else            v = pr[c - 3];  // pts[c-6] lives at raw col 3+(c-6)
    }
    A0[e] = (_Float16)v;
  }
  __syncthreads();

  mlp_layer(A0, W1s, 64, 32, al1, bt1, A1, wave, lane);  __syncthreads();
  mlp_layer(A1, W2s, 32, 48, al2, bt2, A2, wave, lane);  __syncthreads();
  mlp_layer(A2, W3s, 64, 64, al3, bt3, A1, wave, lane);  __syncthreads();

  // attention logits over 32 points
  if (tid < 32) {
    float s = p.ab[0];
    for (int f = 0; f < 64; ++f) s += (float)A1[tid * 64 + f] * wat[f];
    lg[tid] = s;
  }
  __syncthreads();
  if (tid < 32) {
    float mx = -3.4e38f;
    for (int m = 0; m < 32; ++m) mx = fmaxf(mx, lg[m]);
    float sum = 0.f;
    for (int m = 0; m < 32; ++m) sum += __expf(lg[m] - mx);
    const float w = __expf(lg[tid] - mx) / sum;
    wsm[tid] = w;
    attn_out[(size_t)g * NSK + tid] = w;   // output #2
  }
  __syncthreads();
  if (tid < 64) {
    float s = 0.f;
    for (int m = 0; m < 32; ++m) s += (float)A1[m * 64 + tid] * wsm[m];
    voxel[(size_t)g * 64 + tid] = s;
  }
}

// ---------------------------------------------------------------------------
// Kernel 3: conv1 (96,64,3,3,3) VALID on (64,9,3,3) as im2col GEMM:
// per frame A(16x1728 f16, rows 7..15 zero) x B(1728x96) with WMMA; B panel
// is pre-converted f16 in workspace (L2-resident), loaded as b128 pairs.
// ---------------------------------------------------------------------------
struct VXP1 { const float *b, *g, *be, *m, *v; };

__global__ void conv1_kernel(const float* __restrict__ voxel,
                             const _Float16* __restrict__ vx1h, VXP1 p,
                             float* __restrict__ c1)
{
  const int bt   = blockIdx.x;
  const int tid  = threadIdx.x;
  const int wave = tid >> 5, lane = tid & 31;

  __shared__ __align__(32) _Float16 Acol[16 * 1728];
  __shared__ float al[96], be[96];

  if (tid < 96) { const float s = p.g[tid] * rsqrtf(p.v[tid] + EPSBN);
                  al[tid] = s; be[tid] = (p.b[tid] - p.m[tid]) * s + p.be[tid]; }

  // im2col: k = ic*27 + kd*9 + kh*3 + kw, spatial a = (d+kd)*9 + kh*3 + kw
  for (int e = tid; e < 16 * 1728; e += 128) {
    const int m = e / 1728, k = e - m * 1728;
    float v = 0.f;
    if (m < 7) {
      const int ic  = k / 27;
      const int rem = k - ic * 27;
      const int sp  = (m + rem / 9) * 9 + (rem % 9);
      v = voxel[((size_t)bt * NANCH + sp) * 64 + ic];
    }
    Acol[e] = (_Float16)v;
  }
  __syncthreads();

  for (int t = wave; t < 6; t += 4) {              // 6 N-tiles of 16 (96 oc)
    const int oc = t * 16 + (lane & 15);
    const _Float16* arow = Acol + (lane & 15) * 1728;
    const _Float16* wp = vx1h + (size_t)oc * 1728 + ((lane >> 4) << 4);
    v8f acc = {};
#pragma unroll 2
    for (int kc = 0; kc < 1728; kc += 32) {        // 54 K-chunks
      const v16h av = load_afrag(arow, kc, lane);
      const v16h bv = *(const v16h*)(wp + kc);
      acc = __builtin_amdgcn_wmma_f32_16x16x32_f16(false, av, false, bv,
                                                   (short)0, acc, false, false);
    }
    const float a_ = al[oc], b_ = be[oc];
#pragma unroll
    for (int r = 0; r < 8; ++r) {
      const int d = r + ((lane < 16) ? 0 : 8);
      if (d < 7) {
        const float y = acc[r] * a_ + b_;
        c1[((size_t)bt * 96 + oc) * 7 + d] = (y > 0.f ? y : 0.f);
      }
    }
  }
}

// ---------------------------------------------------------------------------
// Kernel 4: conv2 (128,96,5,1,1) + conv3 (64,128,3,1,1), tiny -> plain VALU.
// c1 activations staged via async global->LDS b128.
// ---------------------------------------------------------------------------
struct VXP23 { const float *w2, *b2, *g2, *be2, *m2, *v2;
               const float *w3, *b3, *g3, *be3, *m3, *v3; };

__global__ void conv23_kernel(const float* __restrict__ c1, VXP23 p,
                              float* __restrict__ vv)
{
  const int bt  = blockIdx.x;
  const int tid = threadIdx.x;
  __shared__ __align__(16) float c1l[96 * 7];   // 672 floats = 168 b128 units
  __shared__ float c2l[128 * 3];
  __shared__ float a2[128], b2v[128], a3[64], b3v[64];

  const float* c1g = c1 + (size_t)bt * 672;
  for (int u = tid; u < 168; u += 128) async_b128(c1g + u * 4, c1l + u * 4);
  if (tid < 128) { const float s = p.g2[tid] * rsqrtf(p.v2[tid] + EPSBN);
                   a2[tid] = s; b2v[tid] = (p.b2[tid] - p.m2[tid]) * s + p.be2[tid]; }
  if (tid < 64)  { const float s = p.g3[tid] * rsqrtf(p.v3[tid] + EPSBN);
                   a3[tid] = s; b3v[tid] = (p.b3[tid] - p.m3[tid]) * s + p.be3[tid]; }
  wait_async0();
  __syncthreads();

  for (int e = tid; e < 384; e += 128) {           // 128 oc x 3 d
    const int oc = e / 3, d = e - oc * 3;
    const float* wr = p.w2 + (size_t)oc * 480;
    float s = 0.f;
    for (int ic = 0; ic < 96; ++ic)
#pragma unroll
      for (int kd = 0; kd < 5; ++kd)
        s += c1l[ic * 7 + d + kd] * wr[ic * 5 + kd];
    const float y = s * a2[oc] + b2v[oc];
    c2l[e] = (y > 0.f ? y : 0.f);
  }
  __syncthreads();

  if (tid < 64) {                                  // 64 oc x 1 d
    const float* wr = p.w3 + (size_t)tid * 384;
    float s = 0.f;
    for (int ic = 0; ic < 128; ++ic)
#pragma unroll
      for (int kd = 0; kd < 3; ++kd)
        s += c2l[ic * 3 + kd] * wr[ic * 3 + kd];
    const float y = s * a3[tid] + b3v[tid];
    vv[(size_t)bt * 64 + tid] = (y > 0.f ? y : 0.f);
  }
}

// ---------------------------------------------------------------------------
// Kernel 5: LSTM scan (16 steps, HID=64). Block per batch element, 64 threads.
// ---------------------------------------------------------------------------
__global__ void lstm_kernel(const float* __restrict__ vv,
                            const float* __restrict__ h0,
                            const float* __restrict__ c0,
                            const float* __restrict__ wih,
                            const float* __restrict__ whh,
                            const float* __restrict__ bih,
                            const float* __restrict__ bhh,
                            float* __restrict__ a_out,
                            float* __restrict__ hn_out,
                            float* __restrict__ cn_out)
{
  const int b   = blockIdx.x;
  const int tid = threadIdx.x;  // 64
  __shared__ float h[64], c[64], xs[64], z[256];
  h[tid] = h0[b * 64 + tid];
  c[tid] = c0[b * 64 + tid];
  __syncthreads();
  for (int t = 0; t < 16; ++t) {
    xs[tid] = vv[((size_t)b * 16 + t) * 64 + tid];
    __syncthreads();
    for (int j = tid; j < 256; j += 64) {
      const float* wi = wih + (size_t)j * 64;
      const float* wh = whh + (size_t)j * 64;
      float s = bih[j] + bhh[j];
      for (int k = 0; k < 64; ++k) s += xs[k] * wi[k] + h[k] * wh[k];
      z[j] = s;
    }
    __syncthreads();
    const float ig = 1.f / (1.f + __expf(-z[tid]));
    const float fg = 1.f / (1.f + __expf(-z[64 + tid]));
    const float gg = tanhf(z[128 + tid]);
    const float og = 1.f / (1.f + __expf(-z[192 + tid]));
    const float cn = fg * c[tid] + ig * gg;
    const float hn = og * tanhf(cn);
    __syncthreads();
    c[tid] = cn; h[tid] = hn;
    a_out[((size_t)b * 16 + t) * 64 + tid] = hn;   // output #1
    __syncthreads();
  }
  hn_out[b * 64 + tid] = h[tid];                   // output #3
  cn_out[b * 64 + tid] = c[tid];                   // output #4
}

// ---------------------------------------------------------------------------
// Launcher. d_in order: x, g_loc, h0, c0, 42 params in dict order, 3 scalars.
// d_out: [a_vec 8192 | attn 331776 | hn 512 | cn 512] f32.
// ---------------------------------------------------------------------------
extern "C" void kernel_launch(void* const* d_in, const int* in_sizes, int n_in,
                              void* d_out, int out_size, void* d_ws, size_t ws_size,
                              hipStream_t stream)
{
  (void)in_sizes; (void)n_in; (void)out_size; (void)ws_size;
  const float* x     = (const float*)d_in[0];
  const float* g_loc = (const float*)d_in[1];
  const float* h0    = (const float*)d_in[2];
  const float* c0    = (const float*)d_in[3];
  const float* P[42];
  for (int i = 0; i < 42; ++i) P[i] = (const float*)d_in[4 + i];
  // P: 0..5 pn1, 6..11 pn2, 12..17 pn3, 18 attn_w, 19 attn_b,
  //    20..25 vx1, 26..31 vx2, 32..37 vx3, 38 wih, 39 whh, 40 bih, 41 bhh

  char* ws = (char*)d_ws;
  size_t off = 0;
  int*      idx_ws = (int*)(ws + off);      off += (size_t)NGRP * NSK * sizeof(int);      off = (off + 255) & ~(size_t)255;
  float*    voxel  = (float*)(ws + off);    off += (size_t)NGRP * 64 * sizeof(float);     off = (off + 255) & ~(size_t)255;
  float*    c1ws   = (float*)(ws + off);    off += (size_t)BTF * 96 * 7 * sizeof(float);  off = (off + 255) & ~(size_t)255;
  float*    vvws   = (float*)(ws + off);    off += (size_t)BTF * 64 * sizeof(float);      off = (off + 255) & ~(size_t)255;
  _Float16* w1h    = (_Float16*)(ws + off); off += (size_t)32 * 64 * 2;                   off = (off + 255) & ~(size_t)255;
  _Float16* w2h    = (_Float16*)(ws + off); off += (size_t)48 * 32 * 2;                   off = (off + 255) & ~(size_t)255;
  _Float16* w3h    = (_Float16*)(ws + off); off += (size_t)64 * 64 * 2;                   off = (off + 255) & ~(size_t)255;
  _Float16* vx1h   = (_Float16*)(ws + off);

  float* out    = (float*)d_out;
  float* a_out  = out;                   // (8,16,64)
  float* attn_o = out + 8192;            // (10368,32,1)
  float* hn_o   = out + 8192 + 331776;   // (1,8,64)
  float* cn_o   = hn_o + 512;            // (1,8,64)

  prep_kernel<<<dim3(256), dim3(256), 0, stream>>>(P[0], P[6], P[12], P[20],
                                                   w1h, w2h, w3h, vx1h);

  topk_kernel<<<dim3(NGRP), dim3(256), 0, stream>>>(x, g_loc, idx_ws);

  PNP pn = { P[1],  P[2],  P[3],  P[4],  P[5],
             P[7],  P[8],  P[9],  P[10], P[11],
             P[13], P[14], P[15], P[16], P[17],
             P[18], P[19] };
  pointnet_kernel<<<dim3(NGRP), dim3(128), 0, stream>>>(x, g_loc, idx_ws,
                                                        w1h, w2h, w3h, pn,
                                                        attn_o, voxel);

  VXP1 v1 = { P[21], P[22], P[23], P[24], P[25] };
  conv1_kernel<<<dim3(BTF), dim3(128), 0, stream>>>(voxel, vx1h, v1, c1ws);

  VXP23 v23 = { P[26], P[27], P[28], P[29], P[30], P[31],
                P[32], P[33], P[34], P[35], P[36], P[37] };
  conv23_kernel<<<dim3(BTF), dim3(128), 0, stream>>>(c1ws, v23, vvws);

  lstm_kernel<<<dim3(8), dim3(64), 0, stream>>>(vvws, h0, c0,
                                                P[38], P[39], P[40], P[41],
                                                a_out, hn_o, cn_o);
}
